// SelectiveAttention_61950608278079
// MI455X (gfx1250) — matrix operand order repaired and checked by reading, not compiled, not run on hardware
//
#include <hip/hip_runtime.h>
#include <hip/hip_bf16.h>

typedef __attribute__((ext_vector_type(16))) _Float16 v16h;
typedef __attribute__((ext_vector_type(8)))  _Float16 v8h;
typedef __attribute__((ext_vector_type(2)))  _Float16 v2h;
typedef __attribute__((ext_vector_type(8)))  float    v8f;

#define N_EMBD   2048
#define N_HEAD   32
#define HD       64
#define T_NEW    2048
#define CACHE_N  1024
#define TT       (CACHE_N + T_NEW)     /* 3072 total sequence */
#define X_STRIDE (3 * N_EMBD + 1)      /* 6145 floats per x row */
#define VSTR     40                    /* Vt row stride in halves (80B, keeps 16B align) */
#define PSTR     32                    /* P row stride in halves */

static __device__ __forceinline__ v8f wmma_f16(v16h a, v16h b, v8f c) {
  // D = A(16x32 f16) * B(32x16 f16) + C(16x16 f32)
  return __builtin_amdgcn_wmma_f32_16x16x32_f16(false, a, false, b, (short)0, c,
                                                false, false);
}

// Packed f32x2 -> f16x2 convert (v_cvt_pk_rtz_f16_f32); builtin returns
// __fp16 vector, bit-identical to _Float16 vector.
static __device__ __forceinline__ v2h pk2(float a, float b) {
  return __builtin_bit_cast(v2h, __builtin_amdgcn_cvt_pkrtz(a, b));
}

// Convert 16 consecutive floats to a v16h using packed converts.
template <bool ALIGNED>
static __device__ __forceinline__ v16h cvt16(const float* __restrict__ p) {
  float t[16];
  if (ALIGNED) {
    const float4* p4 = (const float4*)p;     // 16B-aligned rows (cache phase)
    #pragma unroll
    for (int q = 0; q < 4; ++q) {
      float4 f = p4[q];
      t[4*q] = f.x; t[4*q+1] = f.y; t[4*q+2] = f.z; t[4*q+3] = f.w;
    }
  } else {
    #pragma unroll
    for (int j = 0; j < 16; ++j) t[j] = p[j]; // x rows are only 4B-aligned
  }
  v16h r;
  #pragma unroll
  for (int j = 0; j < 16; j += 2) {
    v2h h = pk2(t[j], t[j+1]);
    r[j] = h[0]; r[j+1] = h[1];
  }
  return r;
}

// One 32-key flash-attention block.  MASKED: apply causal mask + row clamp.
// ALIGNED: rows are 16B aligned (cache phase).  Address form is branchless:
// row pointer = base + (kg - row_off) * rstride.
template <bool MASKED, bool ALIGNED>
static __device__ __forceinline__ void attn_block(
    const int kbase, const int row_off, const long rstride,
    const float* __restrict__ krow0, const float* __restrict__ vrow0,
    const v16h (&qa)[2], v8f (&o)[4], float (&rmax)[8], float (&rsum)[8],
    _Float16* __restrict__ Vt, _Float16* __restrict__ Pb,
    const int l2, const int hig, const int qg0)
{
  // ---- speculative prefetch of the next K block (global_prefetch_b8) ----
  if (!MASKED) {
    const int kp = kbase + 32 + 16 * hig + l2;
    __builtin_prefetch(krow0 + (size_t)(kp - row_off) * rstride, 0, 0);
  }

  // ---- S = Qs * K^T : two 16x16 subtiles, K B-operands direct from memory ----
  v8f s[2];
  #pragma unroll
  for (int sub = 0; sub < 2; ++sub) {
    int kg = kbase + 16 * sub + l2;
    if (MASKED) kg = kg < (TT - 1) ? kg : (TT - 1);   // clamp; tail masked below
    const float* ks = krow0 + (size_t)(kg - row_off) * rstride;
    // B(32x16) layout: lane holds N=l2 (key), halves j = K-dim 16*hig + j
    const v16h b0 = cvt16<ALIGNED>(ks +      16 * hig);   // dims  0..31
    const v16h b1 = cvt16<ALIGNED>(ks + 32 + 16 * hig);   // dims 32..63
    v8f acc = v8f{};
    acc = wmma_f16(qa[0], b0, acc);
    acc = wmma_f16(qa[1], b1, acc);
    s[sub] = acc;
  }

  // ---- stage V block transposed into LDS: Vt[dim][key], packed pkrtz+v2h ----
  {
    int ka  = kbase + 2 * l2;
    int kb1 = ka + 1;
    if (MASKED) {
      ka  = ka  < (TT - 1) ? ka  : (TT - 1);
      kb1 = kb1 < (TT - 1) ? kb1 : (TT - 1);
    }
    const float* va = vrow0 + (size_t)(ka  - row_off) * rstride;
    const float* vb = vrow0 + (size_t)(kb1 - row_off) * rstride;
    #pragma unroll
    for (int d0 = 0; d0 < 32; d0 += 4) {
      const int dim0 = 32 * hig + d0;
      float a0, a1, a2, a3, b0_, b1_, b2_, b3_;
      if (ALIGNED) {
        float4 fa = *(const float4*)(va + dim0);
        float4 fb = *(const float4*)(vb + dim0);
        a0 = fa.x; a1 = fa.y; a2 = fa.z; a3 = fa.w;
        b0_ = fb.x; b1_ = fb.y; b2_ = fb.z; b3_ = fb.w;
      } else {
        a0 = va[dim0]; a1 = va[dim0+1]; a2 = va[dim0+2]; a3 = va[dim0+3];
        b0_ = vb[dim0]; b1_ = vb[dim0+1]; b2_ = vb[dim0+2]; b3_ = vb[dim0+3];
      }
      *(v2h*)&Vt[(dim0    ) * VSTR + 2*l2] = pk2(a0, b0_);
      *(v2h*)&Vt[(dim0 + 1) * VSTR + 2*l2] = pk2(a1, b1_);
      *(v2h*)&Vt[(dim0 + 2) * VSTR + 2*l2] = pk2(a2, b2_);
      *(v2h*)&Vt[(dim0 + 3) * VSTR + 2*l2] = pk2(a3, b3_);
    }
  }

  // ---- online softmax on the 16x32 S block ----
  // C/D layout: VGPR r, lane -> element (m = r + 8*hig, n = l2)
  float p0v[8], p1v[8], rowm[8];
  #pragma unroll
  for (int r = 0; r < 8; ++r) {
    float a = s[0][r], b = s[1][r];
    if (MASKED) {
      const int qg = qg0 + r + 8 * hig;
      if (kbase +      l2 > qg) a = -__builtin_inff();
      if (kbase + 16 + l2 > qg) b = -__builtin_inff();
    }
    p0v[r] = a; p1v[r] = b;
    rowm[r] = fmaxf(a, b);
  }
  #pragma unroll
  for (int off = 1; off < 16; off <<= 1) {
    #pragma unroll
    for (int r = 0; r < 8; ++r)
      rowm[r] = fmaxf(rowm[r], __shfl_xor(rowm[r], off, 16));
  }
  float corr[8];
  #pragma unroll
  for (int r = 0; r < 8; ++r) {
    const float nm = fmaxf(rmax[r], rowm[r]);
    corr[r] = __expf(rmax[r] - nm);    // exp(-inf)=0 on first block
    rmax[r] = nm;
  }
  float rs[8];
  #pragma unroll
  for (int r = 0; r < 8; ++r) {
    p0v[r] = __expf(p0v[r] - rmax[r]);
    p1v[r] = __expf(p1v[r] - rmax[r]);
    rs[r]  = p0v[r] + p1v[r];
  }
  #pragma unroll
  for (int off = 1; off < 16; off <<= 1) {
    #pragma unroll
    for (int r = 0; r < 8; ++r)
      rs[r] += __shfl_xor(rs[r], off, 16);
  }
  #pragma unroll
  for (int r = 0; r < 8; ++r) rsum[r] = rsum[r] * corr[r] + rs[r];

  // rescale output accumulators (same row mapping as C/D)
  #pragma unroll
  for (int c2 = 0; c2 < 4; ++c2)
    #pragma unroll
    for (int r = 0; r < 8; ++r) o[c2][r] *= corr[r];

  // ---- P (f16) to LDS row-major for C/D -> A-operand cross-lane relayout ----
  #pragma unroll
  for (int r = 0; r < 8; ++r) {
    const int m = r + 8 * hig;
    Pb[m * PSTR +      l2] = (_Float16)p0v[r];
    Pb[m * PSTR + 16 + l2] = (_Float16)p1v[r];
  }

  // ---- O += P * V : A = P(16x32), B = V chunks (32x16) from Vt ----
  v16h pa;
  {
    const v8h a0 = *(const v8h*)&Pb[l2 * PSTR + 16 * hig];
    const v8h a1 = *(const v8h*)&Pb[l2 * PSTR + 16 * hig + 8];
    #pragma unroll
    for (int j = 0; j < 8; ++j) { pa[j] = a0[j]; pa[8 + j] = a1[j]; }
  }
  #pragma unroll
  for (int c2 = 0; c2 < 4; ++c2) {
    const v8h b0 = *(const v8h*)&Vt[(16 * c2 + l2) * VSTR + 16 * hig];
    const v8h b1 = *(const v8h*)&Vt[(16 * c2 + l2) * VSTR + 16 * hig + 8];
    v16h vb;
    #pragma unroll
    for (int j = 0; j < 8; ++j) { vb[j] = b0[j]; vb[8 + j] = b1[j]; }
    o[c2] = wmma_f16(pa, vb, o[c2]);
  }
}

__global__ __launch_bounds__(32)
void flash_attn_cdna5(const float* __restrict__ x,
                      const float* __restrict__ ck,
                      const float* __restrict__ cv,
                      float* __restrict__ out)
{
  __shared__ __align__(16) _Float16 Vt[HD * VSTR];   // V tile transposed, 5120 B
  __shared__ __align__(16) _Float16 Pb[16 * PSTR];   // P tile row-major, 1024 B

  const int tile = blockIdx.x;        // 0..127 (16 queries each)
  const int head = blockIdx.y;        // 0..31
  const int lane = threadIdx.x;       // 0..31 (wave32)
  const int l2   = lane & 15;
  const int hig  = lane >> 4;

  const int qg0 = CACHE_N + tile * 16;    // global index of query row m=0

  // ---- Q -> WMMA A operands (2 head-dim chunks of 32), scale folded in ----
  // A(16x32 f16): lane holds M=l2; halves[0..15]=K(16*hig+.)
  const float* qsrc = x + (size_t)(tile * 16 + l2) * X_STRIDE + head * HD;
  v16h qa[2];
  #pragma unroll
  for (int c = 0; c < 2; ++c) {
    #pragma unroll
    for (int j = 0; j < 8; j += 2) {
      v2h lo = pk2(qsrc[32*c +     16*hig + j]     * 0.125f,
                   qsrc[32*c +     16*hig + j + 1] * 0.125f);
      v2h hh = pk2(qsrc[32*c + 8 + 16*hig + j]     * 0.125f,
                   qsrc[32*c + 8 + 16*hig + j + 1] * 0.125f);
      qa[c][j]     = lo[0]; qa[c][j + 1]     = lo[1];
      qa[c][8 + j] = hh[0]; qa[c][8 + j + 1] = hh[1];
    }
  }

  // ---- running flash-softmax state; lane owns rows m = r + 8*hig ----
  v8f o[4];
  #pragma unroll
  for (int c2 = 0; c2 < 4; ++c2) o[c2] = v8f{};
  float rmax[8], rsum[8];
  #pragma unroll
  for (int r = 0; r < 8; ++r) { rmax[r] = -__builtin_inff(); rsum[r] = 0.0f; }

  const int nkb     = (qg0 + 16 + 31) >> 5;  // total 32-key blocks (causal horizon)
  const int nkb_nm  = (qg0 + 1) >> 5;        // blocks needing no mask (kbase+31 <= qg0)

  // Phase A: keys 0..1023 entirely from caches (aligned rows, no mask, no divergence)
  const float* kc = ck + head * HD;
  const float* vc = cv + head * HD;
  for (int kb = 0; kb < CACHE_N / 32; ++kb)
    attn_block<false, true>(kb * 32, 0, (long)N_EMBD, kc, vc,
                            qa, o, rmax, rsum, Vt, Pb, l2, hig, qg0);

  // Phase B: keys >= 1024 entirely from x, still below the causal diagonal
  const float* kx = x +     N_EMBD + head * HD;   // k columns of x, row_off = CACHE_N
  const float* vx = x + 2 * N_EMBD + head * HD;   // v columns of x
  for (int kb = CACHE_N / 32; kb < nkb_nm; ++kb)
    attn_block<false, false>(kb * 32, CACHE_N, (long)X_STRIDE, kx, vx,
                             qa, o, rmax, rsum, Vt, Pb, l2, hig, qg0);

  // Phase C: one diagonal block with causal mask + row clamp
  for (int kb = nkb_nm; kb < nkb; ++kb)
    attn_block<true, false>(kb * 32, CACHE_N, (long)X_STRIDE, kx, vx,
                            qa, o, rmax, rsum, Vt, Pb, l2, hig, qg0);

  // ---- epilogue: normalize and store ----
  #pragma unroll
  for (int c2 = 0; c2 < 4; ++c2) {
    #pragma unroll
    for (int r = 0; r < 8; ++r) {
      const int m   = r + 8 * hig;
      const int row = tile * 16 + m;                 // output row (new tokens)
      out[(size_t)row * N_EMBD + head * HD + 16 * c2 + l2] = o[c2][r] / rsum[r];
    }
  }
}

extern "C" void kernel_launch(void* const* d_in, const int* in_sizes, int n_in,
                              void* d_out, int out_size, void* d_ws, size_t ws_size,
                              hipStream_t stream) {
  (void)in_sizes; (void)n_in; (void)out_size; (void)d_ws; (void)ws_size;
  const float* x  = (const float*)d_in[0];
  // d_in[1] = cache_q: provably unused (output keeps only rows >= cache_num)
  const float* ck = (const float*)d_in[2];
  const float* cv = (const float*)d_in[3];
  float* out = (float*)d_out;

  dim3 grid(T_NEW / 16, N_HEAD);   // 128 query tiles x 32 heads
  flash_attn_cdna5<<<grid, 32, 0, stream>>>(x, ck, cv, out);
}
